// Simple_graph_net_48095043780761
// MI455X (gfx1250) — matrix operand order repaired and checked by reading, not compiled, not run on hardware
//
#include <hip/hip_runtime.h>
#include <math.h>
#include <stdint.h>

// ---------------------------------------------------------------------------
// Problem constants (from the reference)
// ---------------------------------------------------------------------------
#define HH    1024
#define WW    1024
#define CIN   8
#define NTOK  16384     // N = 4 * (G/2)^2
#define NE    131072    // E = 8 * N

typedef __attribute__((ext_vector_type(16))) _Float16 v16h;
typedef __attribute__((ext_vector_type(8)))  _Float16 v8h;
typedef __attribute__((ext_vector_type(8)))  float    v8f;
typedef __attribute__((ext_vector_type(4)))  int      v4i;

// ---------------------------------------------------------------------------
// Workspace layout (bytes). Total ~185 MB; all intermediates f16 where hot.
// ---------------------------------------------------------------------------
static constexpr size_t OFF_TOKENS = 0;                                   // [N][16][16][8] f16
static constexpr size_t OFF_T1     = OFF_TOKENS + (size_t)NTOK*256*8*2;   // [N][64][48]    f16
static constexpr size_t OFF_T2     = OFF_T1     + (size_t)NTOK*64*48*2;   // [N][16][48]    f16
static constexpr size_t OFF_TOK6   = OFF_T2     + (size_t)NTOK*16*48*2;   // [N][6]         f32
static constexpr size_t OFF_LABEL  = OFF_TOK6   + (size_t)NTOK*6*4;       // [N]            f32
static constexpr size_t OFF_AGG    = OFF_LABEL  + (size_t)NTOK*4;         // [N][10]        f32
static constexpr size_t OFF_B1     = OFF_AGG    + (size_t)NTOK*10*4;      // packed 3*3*512 f16
static constexpr size_t OFF_B2     = OFF_B1     + (size_t)3*3*512*2;      // packed 12*3*512 f16
static constexpr size_t OFF_B3     = OFF_B2     + (size_t)12*3*512*2;     // packed 20*1*512 f16

// ---------------------------------------------------------------------------
// CDNA5 async global->LDS copy (16B per lane, tracked by ASYNCcnt)
// Builtin signature on this toolchain: (v4i AS1*, v4i AS3*, imm offset, imm cpol)
// ---------------------------------------------------------------------------
__device__ __forceinline__ void async_g2l_b128(void* lds, const void* g) {
#if defined(__has_builtin) && __has_builtin(__builtin_amdgcn_global_load_async_to_lds_b128)
  typedef __attribute__((address_space(1))) v4i* gp_t;
  typedef __attribute__((address_space(3))) v4i* lp_t;
  __builtin_amdgcn_global_load_async_to_lds_b128(
      (gp_t)(uintptr_t)g, (lp_t)(uint32_t)(uintptr_t)lds, 0, 0);
#else
  uint32_t l = (uint32_t)(uintptr_t)lds;   // LDS aperture: addr[31:0] = LDS offset
  asm volatile("global_load_async_to_lds_b128 %0, %1, off"
               :: "v"(l), "v"(g) : "memory");
#endif
}

__device__ __forceinline__ void async_wait0() {
#if defined(__has_builtin) && __has_builtin(__builtin_amdgcn_s_wait_asynccnt)
  __builtin_amdgcn_s_wait_asynccnt(0);
#else
  asm volatile("s_wait_asynccnt 0x0" ::: "memory");
#endif
}

// ---------------------------------------------------------------------------
// WMMA helpers (CDNA5 wave32, v_wmma_f32_16x16x32_f16)
// ---------------------------------------------------------------------------
__device__ __forceinline__ v8f do_wmma(v16h a, v16h b, v8f c) {
  return __builtin_amdgcn_wmma_f32_16x16x32_f16(false, a, false, b, (short)0, c,
                                                false, false);
}

// A fragment from an LDS row-major [16 x >=32] tile (ISA 7.12.2 16-bit A layout)
__device__ __forceinline__ v16h frag_a_lds(const _Float16* t, int ld) {
  int lane = threadIdx.x & 31;
  const _Float16* p = t + (lane & 15) * ld + ((lane & 16) ? 8 : 0);
  v8h lo = *(const v8h*)(p);        // K = k0 .. k0+7
  v8h hi = *(const v8h*)(p + 16);   // K = k0+16 .. k0+23
  return __builtin_shufflevector(lo, hi, 0,1,2,3,4,5,6,7,8,9,10,11,12,13,14,15);
}

// B fragment from a weight matrix pre-packed in fragment order:
// chunk = kc*NT + nt ; each lane owns 16 contiguous halfs (32B global load).
__device__ __forceinline__ v16h frag_b_packed(const _Float16* bpk, int chunk) {
  int lane = threadIdx.x & 31;
  return *(const v16h*)(bpk + ((size_t)chunk * 32 + lane) * 16);
}

__device__ __forceinline__ float relu(float x) { return fmaxf(x, 0.f); }

// ---------------------------------------------------------------------------
// Kernel 0: prepack GEMM weights f32->f16 fragment order; zero the segment-sum
//   entry ((kc*NT+nt)*32 + lane)*16 + e  maps to
//   col = nt*16 + (lane&15), k = kc*32 + (lane&16 ? 8 : 0) + (e&7) + (e&8 ? 16 : 0)
// ---------------------------------------------------------------------------
__global__ void k_prep(const float* __restrict__ w1, const float* __restrict__ w2,
                       const float* __restrict__ wd1,
                       _Float16* __restrict__ B1, _Float16* __restrict__ B2,
                       _Float16* __restrict__ B3, float* __restrict__ agg) {
  int tid = blockIdx.x * blockDim.x + threadIdx.x;
  int stride = gridDim.x * blockDim.x;
  for (int idx = tid; idx < 3 * 3 * 512; idx += stride) {       // conv1 B
    int e = idx & 15, lane = (idx >> 4) & 31, g = idx >> 9;
    int nt = g % 3, kc = g / 3;
    int col = nt * 16 + (lane & 15);
    int k = kc * 32 + ((lane & 16) ? 8 : 0) + (e & 7) + ((e & 8) ? 16 : 0);
    B1[idx] = (k < 72 && col < 40) ? (_Float16)w1[k * 40 + col] : (_Float16)0.f;
  }
  for (int idx = tid; idx < 12 * 3 * 512; idx += stride) {      // conv2 B
    int e = idx & 15, lane = (idx >> 4) & 31, g = idx >> 9;
    int nt = g % 3, kc = g / 3;
    int col = nt * 16 + (lane & 15);
    int k = kc * 32 + ((lane & 16) ? 8 : 0) + (e & 7) + ((e & 8) ? 16 : 0);
    B2[idx] = (k < 360 && col < 40) ? (_Float16)w2[k * 40 + col] : (_Float16)0.f;
  }
  for (int idx = tid; idx < 20 * 512; idx += stride) {          // dense1 B
    int e = idx & 15, lane = (idx >> 4) & 31, kc = idx >> 9;
    int col = lane & 15;
    int k = kc * 32 + ((lane & 16) ? 8 : 0) + (e & 7) + ((e & 8) ? 16 : 0);
    B3[idx] = (col < 6) ? (_Float16)wd1[k * 6 + col] : (_Float16)0.f;
  }
  for (int idx = tid; idx < NTOK * 10; idx += stride) agg[idx] = 0.f;
}

// ---------------------------------------------------------------------------
// Kernel 1: masked token extraction + per-patch label (one block per patch)
// ---------------------------------------------------------------------------
__global__ void __launch_bounds__(256)
k_build_tokens(const float* __restrict__ img, const float* __restrict__ lab,
               const int* __restrict__ masks, _Float16* __restrict__ tokens,
               float* __restrict__ label_svs) {
  __shared__ float s_cnt[256];
  __shared__ float s_lab[256];
  int n = blockIdx.x;
  int s = n >> 12, p = n & 4095;
  int i = p >> 6, j = p & 63;
  int sx = s & 1, sy = s >> 1;                     // SHIFTS: (0,0)(1,0)(0,1)(1,1)
  int idx = sx + 2 * i, idy = sy + 2 * j;
  int t = threadIdx.x;
  int u = t >> 4, v = t & 15;
  int row = (i * 16 + u + sx * 8) & (HH - 1);
  int col = (j * 16 + v + sy * 8) & (WW - 1);
  int mo = (row * WW + col) * 2;
  float fv = (masks[mo] == idx && masks[mo + 1] == idy) ? 1.f : 0.f;
  const float* ip = img + ((size_t)(row * WW + col)) * CIN;
  _Float16* op = tokens + ((size_t)n * 256 + t) * CIN;
#pragma unroll
  for (int c = 0; c < CIN; ++c) op[c] = (_Float16)(ip[c] * fv);

  s_cnt[t] = fv;
  s_lab[t] = lab[row * WW + col] * fv;
  __syncthreads();
  for (int off = 128; off > 0; off >>= 1) {
    if (t < off) { s_cnt[t] += s_cnt[t + off]; s_lab[t] += s_lab[t + off]; }
    __syncthreads();
  }
  if (t == 0) {
    float mean = s_lab[0] / (s_cnt[0] + 1e-8f);
    const float twopi = 6.28318530717958647692f;
    float a = mean - sinf(twopi * mean) / twopi;   // diff_round
    a = a - sinf(twopi * a) / twopi;               // harder_diff_round
    label_svs[n] = a;
  }
}

// ---------------------------------------------------------------------------
// per-pixel channel layernorm + affine + relu, write 48-ch padded f16 row
// ---------------------------------------------------------------------------
__device__ __forceinline__ void ln_relu_store(const float* src, const float* bias,
                                              const float* sc, const float* bi,
                                              _Float16* dst) {
  float y[40], mu = 0.f;
#pragma unroll
  for (int c = 0; c < 40; ++c) { y[c] = src[c] + bias[c]; mu += y[c]; }
  mu *= (1.f / 40.f);
  float var = 0.f;
#pragma unroll
  for (int c = 0; c < 40; ++c) { float d = y[c] - mu; var += d * d; }
  float inv = rsqrtf(var * (1.f / 40.f) + 1e-6f);
#pragma unroll
  for (int c = 0; c < 40; ++c)
    dst[c] = (_Float16)relu((y[c] - mu) * inv * sc[c] + bi[c]);
#pragma unroll
  for (int c = 40; c < 48; ++c) dst[c] = (_Float16)0.f;
}

// ---------------------------------------------------------------------------
// Kernel 2: conv1 as implicit GEMM. One block per token, 4 waves.
// M=64 out pixels, K=72->96 (3 chunks), Nc=40->48 (3 tiles). SAME pad hi only.
// ---------------------------------------------------------------------------
__global__ void __launch_bounds__(128)
k_conv1(const _Float16* __restrict__ tokens, const _Float16* __restrict__ B1,
        const float* __restrict__ bias, const float* __restrict__ sc,
        const float* __restrict__ bi, _Float16* __restrict__ out /*[N][64][48]*/) {
  __shared__ _Float16 sIn[16 * 16 * 8];   // 4 KB
  __shared__ _Float16 sA[64 * 96];        // 12 KB im2col tile
  __shared__ float    sC[64 * 48];        // 12 KB accum
  int n = blockIdx.x, t = threadIdx.x;

  // async-stage the token tile: 256 x 16B straight into LDS (no VGPR bounce)
  {
    const char* g = (const char*)(tokens + (size_t)n * 2048);
    char* l = (char*)sIn;
    for (int k = t; k < 256; k += 128)
      async_g2l_b128(l + k * 16, g + k * 16);
    async_wait0();
  }
  __syncthreads();

  for (int idx = t; idx < 64 * 96; idx += 128) {        // build im2col A
    int r = idx / 96, k = idx - r * 96;
    _Float16 val = (_Float16)0.f;
    if (k < 72) {
      int f = k >> 3, ci = k & 7;
      int du = f / 3, dv = f - du * 3;
      int ix = 2 * (r >> 3) + du, iy = 2 * (r & 7) + dv;
      if (ix < 16 && iy < 16) val = sIn[(ix * 16 + iy) * 8 + ci];
    }
    sA[idx] = val;
  }
  __syncthreads();

  int w = t >> 5, lane = t & 31, colb = lane & 15;
  v8f vz = {0.f,0.f,0.f,0.f,0.f,0.f,0.f,0.f};
  v8f acc[3] = {vz, vz, vz};
  for (int kc = 0; kc < 3; ++kc) {
    v16h a = frag_a_lds(&sA[(w << 4) * 96 + kc * 32], 96);
#pragma unroll
    for (int nt = 0; nt < 3; ++nt)
      acc[nt] = do_wmma(a, frag_b_packed(B1, kc * 3 + nt), acc[nt]);
  }
#pragma unroll
  for (int nt = 0; nt < 3; ++nt)
#pragma unroll
    for (int r = 0; r < 8; ++r) {
      int rowc = (w << 4) + ((lane & 16) ? (r + 8) : r);
      sC[rowc * 48 + (nt << 4) + colb] = acc[nt][r];
    }
  __syncthreads();
  if (t < 64)
    ln_relu_store(&sC[t * 48], bias, sc, bi, out + ((size_t)n * 64 + t) * 48);
}

// ---------------------------------------------------------------------------
// Kernel 3: conv2 implicit GEMM. 4 tokens/block (one per wave), 128 threads.
// M=16 out pixels/token, K=360->384 (12 chunks), A built in registers from LDS.
// ---------------------------------------------------------------------------
__device__ __forceinline__ _Float16 im2col2(const _Float16* in, int ox, int oy, int k) {
  if (k >= 360) return (_Float16)0.f;
  int f = k / 40, ci = k - f * 40;
  int du = f / 3, dv = f - du * 3;
  int ix = 2 * ox + du, iy = 2 * oy + dv;
  if (ix >= 8 || iy >= 8) return (_Float16)0.f;
  return in[(ix * 8 + iy) * 48 + ci];
}

__global__ void __launch_bounds__(128)
k_conv2(const _Float16* __restrict__ t1, const _Float16* __restrict__ B2,
        const float* __restrict__ bias, const float* __restrict__ sc,
        const float* __restrict__ bi, _Float16* __restrict__ out /*[N][16][48]*/) {
  __shared__ _Float16 sIn[4 * 64 * 48];   // 24 KB (4 tokens of conv1 output)
  __shared__ float    sC[4 * 16 * 48];    // 12 KB
  int n0 = blockIdx.x * 4, t = threadIdx.x;

  // async-stage 24 KB of conv1 activations: 1536 x 16B, 12 per thread
  {
    const char* g = (const char*)(t1 + (size_t)n0 * 64 * 48);
    char* l = (char*)sIn;
    for (int k = t; k < 1536; k += 128)
      async_g2l_b128(l + k * 16, g + k * 16);
    async_wait0();
  }
  __syncthreads();

  int w = t >> 5, lane = t & 31, colb = lane & 15;
  const _Float16* in = sIn + w * 64 * 48;
  int row = lane & 15, ox = row >> 2, oy = row & 3;
  int kb = (lane & 16) ? 8 : 0;
  v8f vz = {0.f,0.f,0.f,0.f,0.f,0.f,0.f,0.f};
  v8f acc[3] = {vz, vz, vz};
  for (int kc = 0; kc < 12; ++kc) {
    v16h a;
#pragma unroll
    for (int e = 0; e < 8; ++e) {
      a[e]     = im2col2(in, ox, oy, kc * 32 + kb + e);
      a[8 + e] = im2col2(in, ox, oy, kc * 32 + kb + 16 + e);
    }
#pragma unroll
    for (int nt = 0; nt < 3; ++nt)
      acc[nt] = do_wmma(a, frag_b_packed(B2, kc * 3 + nt), acc[nt]);
  }
#pragma unroll
  for (int nt = 0; nt < 3; ++nt)
#pragma unroll
    for (int r = 0; r < 8; ++r) {
      int rowc = (w << 4) + ((lane & 16) ? (r + 8) : r);
      sC[rowc * 48 + (nt << 4) + colb] = acc[nt][r];
    }
  __syncthreads();
  if (t < 64) {
    int tok = n0 + (t >> 4), pix = t & 15;
    ln_relu_store(&sC[t * 48], bias, sc, bi, out + ((size_t)tok * 16 + pix) * 48);
  }
}

// ---------------------------------------------------------------------------
// Kernel 4: dense1 (640 -> 6, relu). 8 waves/block, 16 tokens/wave.
// ---------------------------------------------------------------------------
__global__ void __launch_bounds__(256)
k_dense1(const _Float16* __restrict__ t2, const _Float16* __restrict__ B3,
         const float* __restrict__ b1, float* __restrict__ tok6) {
  int t = threadIdx.x, w = t >> 5, lane = t & 31;
  int tok0 = blockIdx.x * 128 + w * 16;
  const _Float16* base = t2 + ((size_t)(tok0 + (lane & 15))) * 16 * 48;
  int kb = (lane & 16) ? 8 : 0;
  v8f acc = {0.f,0.f,0.f,0.f,0.f,0.f,0.f,0.f};
  for (int kc = 0; kc < 20; ++kc) {
    v16h a;
#pragma unroll
    for (int e = 0; e < 8; ++e) {
      int k1 = kc * 32 + kb + e, k2 = k1 + 16;
      int p1 = k1 / 40, c1 = k1 - p1 * 40;
      int p2 = k2 / 40, c2 = k2 - p2 * 40;
      a[e]     = base[p1 * 48 + c1];
      a[8 + e] = base[p2 * 48 + c2];
    }
    acc = do_wmma(a, frag_b_packed(B3, kc), acc);
  }
  int col = lane & 15;
  if (col < 6) {
    float bb = b1[col];
#pragma unroll
    for (int r = 0; r < 8; ++r) {
      int tok = tok0 + ((lane & 16) ? (r + 8) : r);
      tok6[tok * 6 + col] = relu(acc[r] + bb);
    }
  }
}

// ---------------------------------------------------------------------------
// Tiny 4-layer MLP (IN ->5 ->5 ->5 ->10), scalar per thread
// ---------------------------------------------------------------------------
template <int IN>
__device__ __forceinline__ void mlp4(const float* x,
    const float* W0, const float* b0, const float* W1, const float* b1,
    const float* W2, const float* b2, const float* W3, const float* b3,
    float* out10) {
  float h1[5], h2[5], h3[5];
#pragma unroll
  for (int j = 0; j < 5; ++j) {
    float a = b0[j];
    for (int i = 0; i < IN; ++i) a += x[i] * W0[i * 5 + j];
    h1[j] = relu(a);
  }
#pragma unroll
  for (int j = 0; j < 5; ++j) {
    float a = b1[j];
#pragma unroll
    for (int i = 0; i < 5; ++i) a += h1[i] * W1[i * 5 + j];
    h2[j] = relu(a);
  }
#pragma unroll
  for (int j = 0; j < 5; ++j) {
    float a = b2[j];
#pragma unroll
    for (int i = 0; i < 5; ++i) a += h2[i] * W2[i * 5 + j];
    h3[j] = relu(a);
  }
#pragma unroll
  for (int j = 0; j < 10; ++j) {
    float a = b3[j];
#pragma unroll
    for (int i = 0; i < 5; ++i) a += h3[i] * W3[i * 10 + j];
    out10[j] = a;
  }
}

// Kernel 5: edge MLP + segment-sum via native fp32 global atomics
__global__ void __launch_bounds__(256)
k_edge(const int* __restrict__ ep, const float* __restrict__ tok6,
       const float* W0, const float* b0, const float* W1, const float* b1,
       const float* W2, const float* b2, const float* W3, const float* b3,
       float* __restrict__ agg) {
  int e = blockIdx.x * blockDim.x + threadIdx.x;
  if (e >= NE) return;
  int s = ep[2 * e], r = ep[2 * e + 1];
  float x[13];
#pragma unroll
  for (int i = 0; i < 6; ++i) { x[i] = tok6[s * 6 + i]; x[6 + i] = tok6[r * 6 + i]; }
  x[12] = 1.f;
  float o[10];
  mlp4<13>(x, W0, b0, W1, b1, W2, b2, W3, b3, o);
#pragma unroll
  for (int j = 0; j < 10; ++j) unsafeAtomicAdd(&agg[r * 10 + j], o[j]);
}

// Kernel 6: node MLP + dense2 + log-softmax CE loss; writes final outputs
__global__ void __launch_bounds__(256)
k_node(const float* __restrict__ tok6, const float* __restrict__ agg,
       const float* __restrict__ label_svs,
       const float* W0, const float* b0, const float* W1, const float* b1,
       const float* W2, const float* b2, const float* W3, const float* b3,
       const float* Wd2, const float* bd2, float* __restrict__ out) {
  int n = blockIdx.x * blockDim.x + threadIdx.x;
  if (n >= NTOK) return;
  float x[17];
#pragma unroll
  for (int i = 0; i < 6; ++i)  x[i] = tok6[n * 6 + i];
#pragma unroll
  for (int i = 0; i < 10; ++i) x[6 + i] = agg[n * 10 + i];
  x[16] = 1.f;
  float h[10];
  mlp4<17>(x, W0, b0, W1, b1, W2, b2, W3, b3, h);
  float l0 = bd2[0], l1 = bd2[1];
#pragma unroll
  for (int i = 0; i < 10; ++i) { l0 += h[i] * Wd2[i * 2]; l1 += h[i] * Wd2[i * 2 + 1]; }
  float y = label_svs[n];
  float m = fmaxf(l0, l1);
  float lse = m + logf(expf(l0 - m) + expf(l1 - m));
  float loss = -((1.f - y) * (l0 - lse) + y * (l1 - lse));
  out[n] = loss;
  out[NTOK + 2 * n]     = l0;
  out[NTOK + 2 * n + 1] = l1;
  out[3 * NTOK + 2 * n]     = 1.f - y;
  out[3 * NTOK + 2 * n + 1] = y;
}

// ---------------------------------------------------------------------------
// Launcher. d_in order (setup_inputs dict order; params pytree alphabetical):
//  0 curr_image 1 curr_label 2 masks 3 edge_pairs
//  4-7 conv1(w,b,sc,bi) 8-11 conv2(w,b,sc,bi) 12-13 dense1(W,b) 14-15 dense2(W,b)
//  16-23 edge_mlp (W,b)x4   24-31 node_mlp (W,b)x4
// ---------------------------------------------------------------------------
extern "C" void kernel_launch(void* const* d_in, const int* in_sizes, int n_in,
                              void* d_out, int out_size, void* d_ws, size_t ws_size,
                              hipStream_t stream) {
  const float* img   = (const float*)d_in[0];
  const float* lab   = (const float*)d_in[1];
  const int*   masks = (const int*)d_in[2];
  const int*   edges = (const int*)d_in[3];
  const float* c1w = (const float*)d_in[4],  *c1b = (const float*)d_in[5];
  const float* c1s = (const float*)d_in[6],  *c1i = (const float*)d_in[7];
  const float* c2w = (const float*)d_in[8],  *c2b = (const float*)d_in[9];
  const float* c2s = (const float*)d_in[10], *c2i = (const float*)d_in[11];
  const float* d1w = (const float*)d_in[12], *d1b = (const float*)d_in[13];
  const float* d2w = (const float*)d_in[14], *d2b = (const float*)d_in[15];
  const float* eW[4] = {(const float*)d_in[16], (const float*)d_in[18],
                        (const float*)d_in[20], (const float*)d_in[22]};
  const float* eB[4] = {(const float*)d_in[17], (const float*)d_in[19],
                        (const float*)d_in[21], (const float*)d_in[23]};
  const float* nW[4] = {(const float*)d_in[24], (const float*)d_in[26],
                        (const float*)d_in[28], (const float*)d_in[30]};
  const float* nB[4] = {(const float*)d_in[25], (const float*)d_in[27],
                        (const float*)d_in[29], (const float*)d_in[31]};

  char* ws = (char*)d_ws;
  _Float16* tokens = (_Float16*)(ws + OFF_TOKENS);
  _Float16* t1     = (_Float16*)(ws + OFF_T1);
  _Float16* t2     = (_Float16*)(ws + OFF_T2);
  float*    tok6   = (float*)(ws + OFF_TOK6);
  float*    lbl    = (float*)(ws + OFF_LABEL);
  float*    agg    = (float*)(ws + OFF_AGG);
  _Float16* B1     = (_Float16*)(ws + OFF_B1);
  _Float16* B2     = (_Float16*)(ws + OFF_B2);
  _Float16* B3     = (_Float16*)(ws + OFF_B3);

  k_prep<<<128, 256, 0, stream>>>(c1w, c2w, d1w, B1, B2, B3, agg);
  k_build_tokens<<<NTOK, 256, 0, stream>>>(img, lab, masks, tokens, lbl);
  k_conv1<<<NTOK, 128, 0, stream>>>(tokens, B1, c1b, c1s, c1i, t1);
  k_conv2<<<NTOK / 4, 128, 0, stream>>>(t1, B2, c2b, c2s, c2i, t2);
  k_dense1<<<NTOK / 128, 256, 0, stream>>>(t2, B3, d1b, tok6);
  k_edge<<<NE / 256, 256, 0, stream>>>(edges, tok6,
      eW[0], eB[0], eW[1], eB[1], eW[2], eB[2], eW[3], eB[3], agg);
  k_node<<<NTOK / 256, 256, 0, stream>>>(tok6, agg, lbl,
      nW[0], nB[0], nW[1], nB[1], nW[2], nB[2], nW[3], nB[3],
      d2w, d2b, (float*)d_out);
}
